// HiPPOCell_57964878626842
// MI455X (gfx1250) — matrix operand order repaired and checked by reading, not compiled, not run on hardware
//
#include <hip/hip_runtime.h>
#include <hip/hip_bf16.h>
#include <stdint.h>

// ---------------------------------------------------------------------------
// HiPPO cell on gfx1250: bf16 WMMA everywhere, chunked (blocked) linear scan.
// Dims: B=8, T=4096, D=512, N=256. Chunk length L=64 -> C=64 chunks.
// ---------------------------------------------------------------------------

typedef __attribute__((ext_vector_type(16))) __bf16 v16bf;
typedef __attribute__((ext_vector_type(8)))  float  v8f;

#define DEVINL static __device__ __forceinline__

constexpr int BSZ = 8;
constexpr int TT  = 4096;
constexpr int DD  = 512;
constexpr int NS  = 256;
constexpr int LCH = 64;          // chunk length
constexpr int NCH = TT / LCH;    // 64 chunks
constexpr int ABS = 264;         // LDS row stride (halves) for A matrix tile
constexpr int SBS = 272;         // LDS row stride (halves) for state buffer

DEVINL __bf16 f2bf(float f) {
  unsigned u = __builtin_bit_cast(unsigned, f);
  u += 0x7fffu + ((u >> 16) & 1u);               // round-to-nearest-even
  unsigned short s = (unsigned short)(u >> 16);
  return __builtin_bit_cast(__bf16, s);
}

DEVINL v8f vzero() {
  v8f z;
#pragma unroll
  for (int i = 0; i < 8; i++) z[i] = 0.0f;
  return z;
}

union Frag16 { v16bf v; uint4 q[2]; };
union Pack16 { __bf16 h[16]; uint4 q[2]; };

// ---------------------------------------------------------------------------
// 128 x (16*NT)-tile bf16 GEMM: C[row, col] = sum_k A[row, k] * Bop[col, k]
// A: row-major, K contiguous (fp32 or bf16, converted during LDS staging).
// Bop: row-major [Ncols][K] bf16 -> B fragments load directly from global
//      (weights are tiny and live in WGP$/L2).
// Optional per-column scale; fp32 and/or bf16 output.
// Requires: M % 128 == 0 (grid.y = M/128), grid.x = Ncols/(16*NT), K % 32 == 0.
// ---------------------------------------------------------------------------
template <bool AF32, int NT>
__global__ __launch_bounds__(256) void k_gemm(
    const void* __restrict__ Ap, const __bf16* __restrict__ Bop,
    float* __restrict__ Cf, __bf16* __restrict__ Cb,
    const float* __restrict__ scale,
    int K, int lda, int ldb, int ldc) {
  __shared__ __bf16 sA[128 * 40];  // 128 rows x 32 K (+8 pad) bf16
  const int tid  = threadIdx.x;
  const int wave = tid >> 5, lane = tid & 31;
  const bool hi  = lane >= 16;
  const int m15  = lane & 15;
  const int mblk = blockIdx.y * 128;
  const int n0   = blockIdx.x * (16 * NT);

  v8f acc[NT];
#pragma unroll
  for (int i = 0; i < NT; i++) acc[i] = vzero();

  const int srow = tid >> 1, shalf = tid & 1;  // 128 rows, 2x16 halves each

  for (int k0 = 0; k0 < K; k0 += 32) {
    // ---- stage A tile (convert fp32 -> bf16 if needed) ----
    Pack16 pk;
    if (AF32) {
      const float* src = (const float*)Ap + (size_t)(mblk + srow) * lda + k0 + shalf * 16;
      if (k0 + 32 < K) __builtin_prefetch(src + 32, 0, 1);  // next K-slice
#pragma unroll
      for (int i = 0; i < 4; i++) {
        float4 f = *(const float4*)(src + i * 4);
        pk.h[i * 4 + 0] = f2bf(f.x);
        pk.h[i * 4 + 1] = f2bf(f.y);
        pk.h[i * 4 + 2] = f2bf(f.z);
        pk.h[i * 4 + 3] = f2bf(f.w);
      }
    } else {
      const __bf16* src = (const __bf16*)Ap + (size_t)(mblk + srow) * lda + k0 + shalf * 16;
      if (k0 + 32 < K) __builtin_prefetch(src + 32, 0, 1);
      pk.q[0] = ((const uint4*)src)[0];
      pk.q[1] = ((const uint4*)src)[1];
    }
    {
      uint4* dst = (uint4*)&sA[srow * 40 + shalf * 16];
      dst[0] = pk.q[0];
      dst[1] = pk.q[1];
    }
    __syncthreads();

    // ---- A fragment: lane = row m, two 8-half runs (K layout of 16b A) ----
    Frag16 fa;
    const int kb = hi ? 8 : 0;
    const __bf16* ap = &sA[(wave * 16 + m15) * 40 + kb];
    fa.q[0] = *(const uint4*)ap;
    fa.q[1] = *(const uint4*)(ap + 16);

#pragma unroll
    for (int nt = 0; nt < NT; nt++) {
      const int j = n0 + nt * 16 + m15;  // output column = Bop row
      const __bf16* bp = Bop + (size_t)j * ldb + k0 + (hi ? 8 : 0);
      Frag16 fb;
      fb.q[0] = *(const uint4*)bp;
      fb.q[1] = *(const uint4*)(bp + 16);
      acc[nt] = __builtin_amdgcn_wmma_f32_16x16x32_bf16(
          false, fa.v, false, fb.v, (short)0, acc[nt], false, false);
    }
    __syncthreads();
  }

  // ---- epilogue ----
#pragma unroll
  for (int nt = 0; nt < NT; nt++) {
    const int col = n0 + nt * 16 + m15;
    const float sc = scale ? scale[col] : 1.0f;
#pragma unroll
    for (int r = 0; r < 8; r++) {
      const int row = mblk + wave * 16 + (hi ? 8 + r : r);
      float v = acc[nt][r] * sc;
      if (Cf) Cf[(size_t)row * ldc + col] = v;
      if (Cb) Cb[(size_t)row * ldc + col] = f2bf(v);
    }
  }
}

// ---------------------------------------------------------------------------
// fp32 -> bf16 convert / transpose-convert (weight prep)
// ---------------------------------------------------------------------------
__global__ void k_convert(const float* __restrict__ src, __bf16* __restrict__ dst, int n) {
  int i = blockIdx.x * 256 + threadIdx.x;
  if (i < n) dst[i] = f2bf(src[i]);
}

__global__ void k_transpose_conv(const float* __restrict__ src, __bf16* __restrict__ dst, int n) {
  int i = blockIdx.x * 256 + threadIdx.x;
  if (i < n * n) {
    int r = i / n, c = i % n;
    dst[c * n + r] = f2bf(src[i]);
  }
}

// ---------------------------------------------------------------------------
// Chunked scan. One workgroup = 2 chunks (16 state rows = 2 chunks x 8 batch).
// A_bar (bf16, 135 KB) lives in LDS; state round-trips through an LDS buffer
// so every wave can read the full K=256 each step.
// MODE 0: local scan, zero seed, emit only chunk-final state (fp32) to Sc.
// MODE 1: true scan, seeded with Sstart, emit every state (bf16) to statesBf
//         via a cooperative, fully-coalesced b128 copy out of the LDS buffer.
// ---------------------------------------------------------------------------
template <int MODE>
__global__ __launch_bounds__(256) void k_scan(
    const __bf16* __restrict__ Abf, const float* __restrict__ X,
    const float* __restrict__ Sstart, float* __restrict__ Sc,
    __bf16* __restrict__ statesBf) {
  extern __shared__ __bf16 lds[];
  __bf16* abar = lds;               // 256 x ABS
  __bf16* sbuf = lds + 256 * ABS;   // 16 x SBS

  const int tid  = threadIdx.x;
  const int wave = tid >> 5, lane = tid & 31;
  const bool hi  = lane >= 16;
  const int m15  = lane & 15;
  const int c0   = blockIdx.x * 2;

  // load A_bar into LDS (one row per thread)
  {
    const uint4* src = (const uint4*)(Abf + tid * NS);
    uint4* dst = (uint4*)(abar + tid * ABS);
    for (int i = 0; i < NS / 8; i++) dst[i] = src[i];
  }
  // per-thread copy lane for the cooperative state<->global moves
  const int crow  = tid >> 4;         // 0..15
  const int ccolb = (tid & 15) * 16;  // 16 halves each
  const int cchunk = c0 + (crow >> 3), cb = crow & 7;

  // init state buffer
  {
    for (int j = 0; j < 16; j++) {
      float v = 0.0f;
      if (MODE == 1) v = Sstart[(cchunk * BSZ + cb) * NS + ccolb + j];
      sbuf[crow * SBS + ccolb + j] = f2bf(v);
    }
  }
  __syncthreads();

  for (int t = 0; t < LCH; t++) {
    v8f acc[2];
    // acc <- x_t  (C operand of the WMMA chain)
#pragma unroll
    for (int nt = 0; nt < 2; nt++) {
      const int col = (wave * 2 + nt) * 16 + m15;
#pragma unroll
      for (int r = 0; r < 8; r++) {
        const int row = hi ? 8 + r : r;
        const int chunk = c0 + (row >> 3), b = row & 7;
        const int tg = chunk * LCH + t;
        acc[nt][r] = X[((size_t)(b * TT + tg)) * NS + col];
      }
    }
    // s_new = s @ A_barT + x : K loop over 256 in steps of 32
#pragma unroll
    for (int kk = 0; kk < NS / 32; kk++) {
      Frag16 fa;
      const int kb = hi ? 8 : 0;
      const __bf16* ap = sbuf + m15 * SBS + kk * 32 + kb;
      fa.q[0] = *(const uint4*)ap;
      fa.q[1] = *(const uint4*)(ap + 16);
#pragma unroll
      for (int nt = 0; nt < 2; nt++) {
        const int j = (wave * 2 + nt) * 16 + m15;  // A_bar row = output column
        const __bf16* bp = abar + j * ABS + kk * 32 + (hi ? 8 : 0);
        Frag16 fb;
        fb.q[0] = *(const uint4*)bp;
        fb.q[1] = *(const uint4*)(bp + 16);
        acc[nt] = __builtin_amdgcn_wmma_f32_16x16x32_bf16(
            false, fa.v, false, fb.v, (short)0, acc[nt], false, false);
      }
    }
    __syncthreads();  // all reads of old sbuf done
#pragma unroll
    for (int nt = 0; nt < 2; nt++) {
      const int col = (wave * 2 + nt) * 16 + m15;
#pragma unroll
      for (int r = 0; r < 8; r++) {
        const int row = hi ? 8 + r : r;
        sbuf[row * SBS + col] = f2bf(acc[nt][r]);
      }
    }
    __syncthreads();
    if (MODE == 1) {
      // cooperative, coalesced copy of the new state out of LDS:
      // 2x ds_load_b128 + 2x global_store_b128 per thread.
      const int tg = cchunk * LCH + t;
      const uint4* s = (const uint4*)&sbuf[crow * SBS + ccolb];
      uint4* g = (uint4*)&statesBf[((size_t)(cb * TT + tg)) * NS + ccolb];
      g[0] = s[0];
      g[1] = s[1];
      // safe: next write to sbuf happens only after the next step's
      // post-K-loop barrier, which orders it after these reads.
    }
    if (MODE == 0 && t == LCH - 1) {
#pragma unroll
      for (int nt = 0; nt < 2; nt++) {
        const int col = (wave * 2 + nt) * 16 + m15;
#pragma unroll
        for (int r = 0; r < 8; r++) {
          const int row = hi ? 8 + r : r;
          const int chunk = c0 + (row >> 3), b = row & 7;
          Sc[(chunk * BSZ + b) * NS + col] = acc[nt][r];
        }
      }
    }
  }
}

// ---------------------------------------------------------------------------
// Carry propagation across the 64 chunks (single WG, 64 serial WMMA steps):
//   Sstart[0]   = state0
//   Sstart[c+1] = Sstart[c] @ (A^L)^T + Sc[c]
//   final_state = Sstart[C]  (written to d_out tail)
// ---------------------------------------------------------------------------
__global__ __launch_bounds__(256) void k_scan_carry(
    const __bf16* __restrict__ Apow, const float* __restrict__ state0,
    const float* __restrict__ Sc, float* __restrict__ Sstart,
    float* __restrict__ finalOut) {
  extern __shared__ __bf16 lds[];
  __bf16* abar = lds;
  __bf16* sbuf = lds + 256 * ABS;

  const int tid  = threadIdx.x;
  const int wave = tid >> 5, lane = tid & 31;
  const bool hi  = lane >= 16;
  const int m15  = lane & 15;

  {
    const uint4* src = (const uint4*)(Apow + tid * NS);
    uint4* dst = (uint4*)(abar + tid * ABS);
    for (int i = 0; i < NS / 8; i++) dst[i] = src[i];
  }
  {
    int row = tid >> 4;
    int colb = (tid & 15) * 16;
    for (int j = 0; j < 16; j++) {
      float v = (row < BSZ) ? state0[row * NS + colb + j] : 0.0f;
      sbuf[row * SBS + colb + j] = f2bf(v);
    }
  }
  for (int i = tid; i < BSZ * NS; i += 256) Sstart[i] = state0[i];  // Sstart[0]
  __syncthreads();

  for (int c = 0; c < NCH; c++) {
    v8f acc[2];
#pragma unroll
    for (int nt = 0; nt < 2; nt++) {
      const int col = (wave * 2 + nt) * 16 + m15;
#pragma unroll
      for (int r = 0; r < 8; r++) {
        const int row = hi ? 8 + r : r;
        acc[nt][r] = (row < BSZ) ? Sc[(c * BSZ + row) * NS + col] : 0.0f;
      }
    }
#pragma unroll
    for (int kk = 0; kk < NS / 32; kk++) {
      Frag16 fa;
      const int kb = hi ? 8 : 0;
      const __bf16* ap = sbuf + m15 * SBS + kk * 32 + kb;
      fa.q[0] = *(const uint4*)ap;
      fa.q[1] = *(const uint4*)(ap + 16);
#pragma unroll
      for (int nt = 0; nt < 2; nt++) {
        const int j = (wave * 2 + nt) * 16 + m15;
        const __bf16* bp = abar + j * ABS + kk * 32 + (hi ? 8 : 0);
        Frag16 fb;
        fb.q[0] = *(const uint4*)bp;
        fb.q[1] = *(const uint4*)(bp + 16);
        acc[nt] = __builtin_amdgcn_wmma_f32_16x16x32_bf16(
            false, fa.v, false, fb.v, (short)0, acc[nt], false, false);
      }
    }
    __syncthreads();
#pragma unroll
    for (int nt = 0; nt < 2; nt++) {
      const int col = (wave * 2 + nt) * 16 + m15;
#pragma unroll
      for (int r = 0; r < 8; r++) {
        const int row = hi ? 8 + r : r;
        sbuf[row * SBS + col] = f2bf(acc[nt][r]);
        if (row < BSZ) {
          if (c + 1 < NCH)
            Sstart[((c + 1) * BSZ + row) * NS + col] = acc[nt][r];
          else
            finalOut[row * NS + col] = acc[nt][r];
        }
      }
    }
    __syncthreads();
  }
}

// ---------------------------------------------------------------------------
// Host-side orchestration
// ---------------------------------------------------------------------------
extern "C" void kernel_launch(void* const* d_in, const int* in_sizes, int n_in,
                              void* d_out, int out_size, void* d_ws, size_t ws_size,
                              hipStream_t stream) {
  (void)in_sizes; (void)n_in; (void)out_size; (void)ws_size;
  const float* u     = (const float*)d_in[0];  // [8,4096,512]
  const float* state = (const float*)d_in[1];  // [8,256]
  const float* W_in  = (const float*)d_in[2];  // [256,512]
  const float* W_out = (const float*)d_in[3];  // [512,256]
  const float* A_bar = (const float*)d_in[4];  // [256,256]
  const float* B_bar = (const float*)d_in[5];  // [256]
  float* out = (float*)d_out;                  // y [8,4096,512] ++ final [8,256]

  char* w = (char*)d_ws;
  auto alloc = [&](size_t bytes) -> char* {
    char* p = w;
    w += (bytes + 255) & ~(size_t)255;
    return p;
  };
  __bf16* Win_bf  = (__bf16*)alloc((size_t)NS * DD * 2);
  __bf16* Wout_bf = (__bf16*)alloc((size_t)DD * NS * 2);
  __bf16* A_bf    = (__bf16*)alloc((size_t)NS * NS * 2);
  __bf16* AT_bf   = (__bf16*)alloc((size_t)NS * NS * 2);
  __bf16* P[2]    = {(__bf16*)alloc((size_t)NS * NS * 2), (__bf16*)alloc((size_t)NS * NS * 2)};
  __bf16* PT[2]   = {(__bf16*)alloc((size_t)NS * NS * 2), (__bf16*)alloc((size_t)NS * NS * 2)};
  float*  X       = (float*)alloc((size_t)BSZ * TT * NS * 4);   // 33.5 MB
  __bf16* Sbf     = (__bf16*)alloc((size_t)BSZ * TT * NS * 2);  // 16.8 MB
  float*  Sc      = (float*)alloc((size_t)NCH * BSZ * NS * 4);
  float*  Sst     = (float*)alloc((size_t)NCH * BSZ * NS * 4);

  // 1) weight prep (fp32 -> bf16, plus A^T for the squaring chain)
  k_convert<<<(NS * DD + 255) / 256, 256, 0, stream>>>(W_in, Win_bf, NS * DD);
  k_convert<<<(DD * NS + 255) / 256, 256, 0, stream>>>(W_out, Wout_bf, DD * NS);
  k_convert<<<(NS * NS + 255) / 256, 256, 0, stream>>>(A_bar, A_bf, NS * NS);
  k_transpose_conv<<<(NS * NS + 255) / 256, 256, 0, stream>>>(A_bar, AT_bf, NS);

  // 2) X = (u @ W_in^T) * B_bar  : M=32768, K=512, Ncols=256, 128x128 tiles
  k_gemm<true, 8><<<dim3(NS / 128, (BSZ * TT) / 128), 256, 0, stream>>>(
      u, Win_bf, X, (__bf16*)nullptr, B_bar, DD, DD, DD, NS);

  // 3) A^64 by 6 squarings (keep both A^(2^i) and its transpose)
  const __bf16* cur = A_bf;
  const __bf16* curT = AT_bf;
  for (int i = 0; i < 6; i++) {
    __bf16* nxt  = P[i & 1];
    __bf16* nxtT = PT[i & 1];
    k_gemm<false, 4><<<dim3(NS / 64, NS / 128), 256, 0, stream>>>(
        cur, curT, (float*)nullptr, nxt, (const float*)nullptr, NS, NS, NS, NS);
    k_gemm<false, 4><<<dim3(NS / 64, NS / 128), 256, 0, stream>>>(
        curT, cur, (float*)nullptr, nxtT, (const float*)nullptr, NS, NS, NS, NS);
    cur = nxt;
    curT = nxtT;
  }

  const size_t ldsbytes = (size_t)(256 * ABS + 16 * SBS) * sizeof(__bf16);  // ~140 KB

  // 4) local scans (zero seed) -> chunk-final states Sc
  k_scan<0><<<NCH / 2, 256, ldsbytes, stream>>>(
      A_bf, X, (const float*)nullptr, Sc, (__bf16*)nullptr);

  // 5) serial carry across chunks with A^64 -> Sstart, final_state
  k_scan_carry<<<1, 256, ldsbytes, stream>>>(
      cur, state, Sc, Sst, out + (size_t)BSZ * TT * DD);

  // 6) seeded re-scan emitting all states (bf16)
  k_scan<1><<<NCH / 2, 256, ldsbytes, stream>>>(
      A_bf, X, Sst, (float*)nullptr, Sbf);

  // 7) y = states @ W_out^T : M=32768, K=256, Ncols=512, 128x128 tiles
  k_gemm<false, 8><<<dim3(DD / 128, (BSZ * TT) / 128), 256, 0, stream>>>(
      Sbf, Wout_bf, out, (__bf16*)nullptr, (const float*)nullptr, NS, NS, NS, DD);
}
